// GATLayer_214748365439
// MI455X (gfx1250) — compile-verified, matrix operand order checked by reading, not run on hardware
//
#include <hip/hip_runtime.h>

typedef __attribute__((ext_vector_type(16))) _Float16 v16h;
typedef __attribute__((ext_vector_type(8)))  float    v8f;

#define N_NODES 50000
#define N_EDGES 800000
#define IN_F    256
#define OUT_F   128
#define HEADS   4
#define DH      32

// Workspace layout (float offsets)
#define WX_OFF     0                                  // N*128 = 6,400,000
#define SSRC_OFF   (N_NODES * OUT_F)                  // + N*4
#define SDST_OFF   (SSRC_OFF + N_NODES * HEADS)       // + N*4
#define DENOM_OFF  (SDST_OFF + N_NODES * HEADS)       // + N*4
#define EEXP_OFF   (DENOM_OFF + N_NODES * HEADS)      // + E*4
#define ACCUM_OFF  (EEXP_OFF + (size_t)N_EDGES * HEADS) // + N*128

__global__ void zero_f32(float* __restrict__ p, int n) {
    int i = blockIdx.x * blockDim.x + threadIdx.x;
    if (i < n) p[i] = 0.0f;
}

// Wx[N,128] = x[N,256] @ W[128,256]^T
// One wave32 per 16x16 output tile. Block = 256 threads = 8 waves; wave w owns
// N-tile w (OUT_F = 8*16). blockIdx.x = M-tile (N_NODES = 3125*16, exact).
// K = 256 consumed in 8 steps of v_wmma_f32_16x16x32_f16 (fp16 in, fp32 accum).
__global__ void __launch_bounds__(256)
gat_gemm_wmma(const float* __restrict__ x, const float* __restrict__ W,
              float* __restrict__ Wx) {
    const int lane     = threadIdx.x & 31;
    const int wave     = threadIdx.x >> 5;
    const int m0       = blockIdx.x * 16;
    const int n0       = wave * 16;
    const int half_sel = lane >> 4;   // ISA 16-bit A/B layout: lanes 16-31 carry K+8
    const int r        = lane & 15;   // A: row M; B: col N

    const float* arow = x + (size_t)(m0 + r) * IN_F;
    const float* brow = W + (size_t)(n0 + r) * IN_F;   // B[k][n] = W[n][k]

    v8f c = {};
    for (int kk = 0; kk < IN_F; kk += 32) {
        const float* ap = arow + kk + half_sel * 8;
        const float* bp = brow + kk + half_sel * 8;
        if (kk + 32 < IN_F) {
            __builtin_prefetch(ap + 32, 0, 3);   // global_prefetch_b8: next K-slab
            __builtin_prefetch(bp + 32, 0, 3);
        }
        v16h a, b;
        #pragma unroll
        for (int i = 0; i < 8; ++i) {
            // lane group 0..15: K = kk + {0..7, 16..23}; group 16..31: +8
            a[i]     = (_Float16)ap[i];
            a[i + 8] = (_Float16)ap[16 + i];
            b[i]     = (_Float16)bp[i];
            b[i + 8] = (_Float16)bp[16 + i];
        }
        c = __builtin_amdgcn_wmma_f32_16x16x32_f16(
                /*neg_a=*/false, a, /*neg_b=*/false, b,
                /*c_mod=*/(short)0, c, /*reuse_a=*/false, /*reuse_b=*/false);
    }
    // D layout: VGPR j -> M = j (+8 for lanes 16-31), N = lane&15
    float* drow = Wx + (size_t)m0 * OUT_F + n0;
    #pragma unroll
    for (int j = 0; j < 8; ++j) {
        int row = j + half_sel * 8;
        drow[(size_t)row * OUT_F + r] = c[j];
    }
}

// s_src[n,h] = dot(Wx[n,h,:], a_w[0:32]);  s_dst[n,h] = dot(Wx[n,h,:], a_w[32:64])
__global__ void gat_scores(const float* __restrict__ Wx,
                           const float* __restrict__ a_w,
                           float* __restrict__ s_src, float* __restrict__ s_dst) {
    int idx = blockIdx.x * blockDim.x + threadIdx.x;
    if (idx >= N_NODES * HEADS) return;
    int n = idx >> 2, h = idx & 3;
    const float* p = Wx + (size_t)n * OUT_F + h * DH;
    float ss = 0.0f, sd = 0.0f;
    #pragma unroll
    for (int d = 0; d < DH; ++d) {
        float v = p[d];
        ss += v * a_w[d];
        sd += v * a_w[DH + d];
    }
    s_src[idx] = ss;
    s_dst[idx] = sd;
}

// Per edge: e_exp = exp(leaky_relu(s_src[src]+s_dst[dst])); denom[dst] += e_exp
__global__ void gat_edge_num(const int* __restrict__ src, const int* __restrict__ dst,
                             const float* __restrict__ s_src, const float* __restrict__ s_dst,
                             float* __restrict__ e_exp, float* __restrict__ denom) {
    int e = blockIdx.x * blockDim.x + threadIdx.x;
    if (e >= N_EDGES) return;
    int s = src[e], d = dst[e];
    #pragma unroll
    for (int h = 0; h < HEADS; ++h) {
        float v = s_src[s * HEADS + h] + s_dst[d * HEADS + h];
        v = v > 0.0f ? v : 0.2f * v;          // leaky_relu(0.2)
        float ex = __expf(v);
        e_exp[(size_t)e * HEADS + h] = ex;
        atomicAdd(&denom[d * HEADS + h], ex); // global_atomic_add_f32, L2-resident
    }
}

// One wave32 per edge: lane = feature-in-head; 4 atomics/lane (one per head).
__global__ void __launch_bounds__(256)
gat_aggregate(const int* __restrict__ src, const int* __restrict__ dst,
              const float* __restrict__ e_exp, const float* __restrict__ denom,
              const float* __restrict__ Wx, float* __restrict__ accum) {
    int gid  = blockIdx.x * blockDim.x + threadIdx.x;
    int e    = gid >> 5;
    int lane = gid & 31;
    if (e >= N_EDGES) return;
    int s = src[e], d = dst[e];
    const float* wsrc = Wx + (size_t)s * OUT_F;
    float*       adst = accum + (size_t)d * OUT_F;
    #pragma unroll
    for (int h = 0; h < HEADS; ++h) {
        float alpha = e_exp[(size_t)e * HEADS + h] / (denom[d * HEADS + h] + 1e-8f);
        atomicAdd(&adst[h * DH + lane], alpha * wsrc[h * DH + lane]);
    }
}

__global__ void gat_elu(const float* __restrict__ accum, float* __restrict__ out) {
    int i = blockIdx.x * blockDim.x + threadIdx.x;
    if (i >= N_NODES * OUT_F) return;
    float v = accum[i];
    out[i] = v > 0.0f ? v : (__expf(v) - 1.0f);
}

extern "C" void kernel_launch(void* const* d_in, const int* in_sizes, int n_in,
                              void* d_out, int out_size, void* d_ws, size_t ws_size,
                              hipStream_t stream) {
    const float* x   = (const float*)d_in[0];
    const int*   ei  = (const int*)d_in[1];     // [2, E] row-major
    const float* W   = (const float*)d_in[2];
    const float* a_w = (const float*)d_in[3];
    const int* src = ei;
    const int* dst = ei + N_EDGES;

    float* ws    = (float*)d_ws;
    float* Wx    = ws + WX_OFF;
    float* s_src = ws + SSRC_OFF;
    float* s_dst = ws + SDST_OFF;
    float* denom = ws + DENOM_OFF;
    float* e_exp = ws + EEXP_OFF;
    float* accum = ws + ACCUM_OFF;
    float* out   = (float*)d_out;

    // Zero the accumulators (harness does not re-poison between replays)
    {
        int n = N_NODES * HEADS;
        zero_f32<<<(n + 255) / 256, 256, 0, stream>>>(denom, n);
        n = N_NODES * OUT_F;
        zero_f32<<<(n + 255) / 256, 256, 0, stream>>>(accum, n);
    }

    // GEMM: 3125 M-tiles x 8 N-tiles, 8 waves/block
    gat_gemm_wmma<<<N_NODES / 16, 256, 0, stream>>>(x, W, Wx);

    // Attention scores
    {
        int n = N_NODES * HEADS;
        gat_scores<<<(n + 127) / 128, 128, 0, stream>>>(Wx, a_w, s_src, s_dst);
    }

    // Softmax numerator + denominator (segment sum via f32 atomics)
    gat_edge_num<<<(N_EDGES + 255) / 256, 256, 0, stream>>>(src, dst, s_src, s_dst,
                                                            e_exp, denom);

    // Weighted scatter: one wave per edge
    {
        long long threads = (long long)N_EDGES * 32;
        gat_aggregate<<<(unsigned)((threads + 255) / 256), 256, 0, stream>>>(
            src, dst, e_exp, denom, Wx, accum);
    }

    // ELU epilogue
    {
        int n = N_NODES * OUT_F;
        gat_elu<<<(n + 255) / 256, 256, 0, stream>>>(accum, out);
    }
}
